// CrossBandBlock_6416681140729
// MI455X (gfx1250) — matrix operand order repaired and checked by reading, not compiled, not run on hardware
//
#include <hip/hip_runtime.h>
#include <math.h>

#define NB 2400      // B*T
#define FF 257
#define CC 64
#define TT 300
#define CPPC 16
#define FP 288       // padded F (K dim and g dim) = 9*32
#define NPAD ((size_t)NB * FP)

typedef _Float16 v16h __attribute__((ext_vector_type(16)));
typedef _Float16 v8h  __attribute__((ext_vector_type(8)));
typedef float    v8f  __attribute__((ext_vector_type(8)));

__device__ __forceinline__ float wave_sum32(float v) {
#pragma unroll
  for (int o = 16; o >= 1; o >>= 1) v += __shfl_xor(v, o, 32);
  return v;
}
// silu via v_exp + v_rcp (no IEEE division chain)
__device__ __forceinline__ float silu_f(float x) {
  return x * __builtin_amdgcn_rcpf(1.f + __expf(-x));
}

// ---------------------------------------------------------------------------
// Prep: fW [16][257][257] f32 -> fW16 [16][288][288] f16 (zero padded),
//       red_w [16][64] -> f16
// ---------------------------------------------------------------------------
__global__ __launch_bounds__(256) void prep_kernel(
    const float* __restrict__ fW, const float* __restrict__ red_w,
    _Float16* __restrict__ fW16, _Float16* __restrict__ redw16) {
  size_t idx = (size_t)blockIdx.x * blockDim.x + threadIdx.x;
  size_t total = (size_t)CPPC * FP * FP;
  if (idx < total) {
    int f = (int)(idx % FP);
    size_t r = idx / FP;
    int g = (int)(r % FP);
    int c = (int)(r / FP);
    float v = (g < FF && f < FF) ? fW[((size_t)c * FF + g) * FF + f] : 0.f;
    fW16[idx] = (_Float16)v;
  }
  if (idx < (size_t)CPPC * CC) redw16[idx] = (_Float16)red_w[idx];
}

// ---------------------------------------------------------------------------
// freq_conv: LN over C, grouped conv along F (G=8, K=3, pad 1), PReLU, residual
// src_is_h / dst_is_h select the [B,F,T,C] <-> [N,F,C] mapping.
// ---------------------------------------------------------------------------
__global__ __launch_bounds__(256) void freq_conv_kernel(
    const float* __restrict__ src, float* __restrict__ dst,
    const float* __restrict__ gam, const float* __restrict__ bet,
    const float* __restrict__ w, const float* __restrict__ cb,
    const float* __restrict__ alphaPtr, int src_is_h, int dst_is_h) {
  __shared__ float yls[66][CC];    // normalized rows f0-1 .. f0+64
  __shared__ float resls[64][CC];  // raw rows f0 .. f0+63
  __shared__ float wls[CC * 8 * 3];

  const int n = blockIdx.x;
  const int f0 = blockIdx.y * 64;
  const int tid = threadIdx.x;
  const int b = n / TT, t = n % TT;
  const float alpha = *alphaPtr;

  for (int i = tid; i < CC * 8 * 3; i += 256) wls[i] = w[i];

  const int wave = tid >> 5, lane = tid & 31;
  for (int idx = wave; idx < 66; idx += 8) {
    const int f = f0 - 1 + idx;
    if (f >= 0 && f < FF) {
      size_t base = src_is_h ? (((size_t)(b * FF + f) * TT + t) * CC)
                             : (((size_t)n * FF + f) * CC);
      float x0 = src[base + lane];
      float x1v = src[base + lane + 32];
      float s  = wave_sum32(x0 + x1v);
      float sq = wave_sum32(x0 * x0 + x1v * x1v);
      float m = s * (1.f / CC);
      float var = sq * (1.f / CC) - m * m;
      float inv = rsqrtf(var + 1e-5f);
      yls[idx][lane]      = (x0 - m) * inv * gam[lane]      + bet[lane];
      yls[idx][lane + 32] = (x1v - m) * inv * gam[lane + 32] + bet[lane + 32];
      const int rr = f - f0;
      if (rr >= 0 && rr < 64) { resls[rr][lane] = x0; resls[rr][lane + 32] = x1v; }
    } else {
      yls[idx][lane] = 0.f;
      yls[idx][lane + 32] = 0.f;
    }
  }
  __syncthreads();

  const int co = tid & 63;
  const int fi = tid >> 6;
  const int g0 = (co >> 3) << 3;
#pragma unroll 1
  for (int i = 0; i < 16; ++i) {
    const int row = fi + 4 * i;
    const int f = f0 + row;
    if (f >= FF) continue;
    float acc = cb[co];
#pragma unroll
    for (int k = 0; k < 3; ++k)
#pragma unroll
      for (int ci = 0; ci < 8; ++ci)
        acc += wls[(co * 8 + ci) * 3 + k] * yls[row + k][g0 + ci];
    const float val = acc >= 0.f ? acc : alpha * acc;
    const float outv = resls[row][co] + val;
    size_t base = dst_is_h ? (((size_t)(b * FF + f) * TT + t) * CC)
                           : (((size_t)n * FF + f) * CC);
    dst[base + co] = outv;
  }
}

// ---------------------------------------------------------------------------
// reduce: hp = silu(LN(x1) @ red_w^T + red_b), stored f16 as hpA[c][n][FP]
// One wave per (n, 16-row f tile); WMMA 16x16x32 f16, K=C=64 (2 steps).
// ---------------------------------------------------------------------------
__global__ __launch_bounds__(32) void reduce_kernel(
    const float* __restrict__ x1, const float* __restrict__ gam,
    const float* __restrict__ bet, const _Float16* __restrict__ redw16,
    const float* __restrict__ red_b, _Float16* __restrict__ hpA) {
  __shared__ _Float16 yls[16][CC];
  const int n = blockIdx.x;
  const int f0 = blockIdx.y * 16;
  const int lane = threadIdx.x;

  for (int m = 0; m < 16; ++m) {
    const int f = f0 + m;
    if (f < FF) {
      size_t base = ((size_t)n * FF + f) * CC;
      float x0 = x1[base + lane];
      float x1v = x1[base + lane + 32];
      float s  = wave_sum32(x0 + x1v);
      float sq = wave_sum32(x0 * x0 + x1v * x1v);
      float mm = s * (1.f / CC);
      float var = sq * (1.f / CC) - mm * mm;
      float inv = rsqrtf(var + 1e-5f);
      yls[m][lane]      = (_Float16)((x0 - mm) * inv * gam[lane]      + bet[lane]);
      yls[m][lane + 32] = (_Float16)((x1v - mm) * inv * gam[lane + 32] + bet[lane + 32]);
    } else {
      yls[m][lane] = (_Float16)0.f;
      yls[m][lane + 32] = (_Float16)0.f;
    }
  }
  __syncthreads();

  const int mr = lane & 15, kh = lane >> 4;
  v8f acc = {0, 0, 0, 0, 0, 0, 0, 0};
#pragma unroll
  for (int kc = 0; kc < 2; ++kc) {
    // A fragment: lane row M=mr; K = kc*32 + kh*16 + 0..15
    v8h alo = *(const v8h*)&yls[mr][kc * 32 + kh * 16];
    v8h ahi = *(const v8h*)&yls[mr][kc * 32 + kh * 16 + 8];
    v16h a = __builtin_shufflevector(alo, ahi, 0, 1, 2, 3, 4, 5, 6, 7, 8, 9, 10,
                                     11, 12, 13, 14, 15);
    // B fragment: lane col N=mr(=p); K = kc*32 + kh*16 + 0..15 (B^T row = red_w row)
    v16h bm = *(const v16h*)&redw16[mr * CC + kc * 32 + kh * 16];
    acc = __builtin_amdgcn_wmma_f32_16x16x32_f16(false, a, false, bm, (short)0,
                                                 acc, false, false);
  }
  // D: lane col p = mr, rows f0 + kh*8 + j
  const int p = mr;
  const float rb = red_b[p];
  v8h st;
#pragma unroll
  for (int j = 0; j < 8; ++j) {
    const int f = f0 + kh * 8 + j;
    const float v = silu_f(acc[j] + rb);
    st[j] = (_Float16)((f < FF) ? v : 0.f);
  }
  *(v8h*)&hpA[(size_t)p * NPAD + (size_t)n * FP + f0 + kh * 8] = st;
}

// ---------------------------------------------------------------------------
// band_gemm: hp2[n,g,c] = sum_f hpA[c][n][f] * fW16[c][g][f] + fB[c][g]
// One wave per (16 n-rows, 32 g-cols, channel c); K = FP = 9 x 32 WMMA steps.
// A tile (16 contiguous rows x 288 f16 = 9KB, one contiguous block) is staged
// into LDS with GLOBAL_LOAD_ASYNC_TO_LDS_B128 (ASYNCcnt), then WMMA A
// fragments come from ds_load_b128; B columns stream from L2-resident fW16.
// ---------------------------------------------------------------------------
__global__ __launch_bounds__(32) void band_gemm_kernel(
    const _Float16* __restrict__ hpA, const _Float16* __restrict__ fW16,
    const float* __restrict__ fB, float* __restrict__ hp2) {
  __shared__ _Float16 aTile[16 * FP];  // 9216 bytes, contiguous copy of A tile

  const int n0 = blockIdx.x * 16;
  const int gp = blockIdx.y;  // 0..8 -> g0 = gp*32
  const int c = blockIdx.z;
  const int lane = threadIdx.x;
  const int li = lane & 15, kh = lane >> 4;

  // --- async stage A tile: hpA[c][n0..n0+15][0..287] is one contiguous block
  {
    const _Float16* Abase = hpA + (size_t)c * NPAD + (size_t)n0 * FP;
    const unsigned ldsBase = (unsigned)(uintptr_t)aTile;  // low 32b = LDS offset
    // 9216 B / (32 lanes * 16 B) = 18 async b128 ops per lane
#pragma unroll
    for (int i = 0; i < 18; ++i) {
      unsigned byteOff = (unsigned)((i * 32 + lane) * 16);
      unsigned ldsAddr = ldsBase + byteOff;
      asm volatile("global_load_async_to_lds_b128 %0, %1, %2 offset:0"
                   :
                   : "v"(ldsAddr), "v"(byteOff), "s"(Abase)
                   : "memory");
    }
    asm volatile("s_wait_asynccnt 0" ::: "memory");
  }
  __syncthreads();

  const _Float16* Arow = aTile + (size_t)li * FP;
  const _Float16* B0 = fW16 + ((size_t)c * FP + (gp * 32 + li)) * FP;
  const _Float16* B1 = B0 + (size_t)16 * FP;

  v8f acc0 = {0, 0, 0, 0, 0, 0, 0, 0};
  v8f acc1 = {0, 0, 0, 0, 0, 0, 0, 0};
  for (int kb = 0; kb < FP; kb += 32) {
    __builtin_prefetch(B0 + kb + 32, 0, 1);
    __builtin_prefetch(B1 + kb + 32, 0, 1);
    v8h alo = *(const v8h*)(Arow + kb + kh * 16);
    v8h ahi = *(const v8h*)(Arow + kb + kh * 16 + 8);
    v16h a = __builtin_shufflevector(alo, ahi, 0, 1, 2, 3, 4, 5, 6, 7, 8, 9, 10,
                                     11, 12, 13, 14, 15);
    v16h b0 = *(const v16h*)(B0 + kb + kh * 16);
    v16h b1 = *(const v16h*)(B1 + kb + kh * 16);
    acc0 = __builtin_amdgcn_wmma_f32_16x16x32_f16(false, a, false, b0, (short)0,
                                                  acc0, false, false);
    acc1 = __builtin_amdgcn_wmma_f32_16x16x32_f16(false, a, false, b1, (short)0,
                                                  acc1, false, false);
  }
  // D: lane col g = g0 + tile*16 + li, rows n0 + kh*8 + j
#pragma unroll
  for (int tile = 0; tile < 2; ++tile) {
    const int g = gp * 32 + tile * 16 + li;
    if (g >= FF) continue;
    const float bias = fB[c * FF + g];
    const v8f acc = tile ? acc1 : acc0;
#pragma unroll
    for (int j = 0; j < 8; ++j) {
      const int nn = n0 + kh * 8 + j;
      hp2[((size_t)nn * FF + g) * CPPC + c] = acc[j] + bias;
    }
  }
}

// ---------------------------------------------------------------------------
// expand: x1 += silu(hp2 @ exp_w^T + exp_b)   (in-place residual)
// ---------------------------------------------------------------------------
__global__ __launch_bounds__(256) void expand_kernel(
    const float* __restrict__ hp2, const float* __restrict__ exp_w,
    const float* __restrict__ exp_b, float* __restrict__ x1) {
  size_t idx = (size_t)blockIdx.x * blockDim.x + threadIdx.x;
  if (idx >= (size_t)NB * FF * CC) return;
  const int cch = (int)(idx % CC);
  const size_t rf = idx / CC;  // n*FF + f
  const float* hrow = hp2 + rf * CPPC;
  float s = exp_b[cch];
#pragma unroll
  for (int p = 0; p < CPPC; ++p) s += hrow[p] * exp_w[cch * CPPC + p];
  x1[idx] += silu_f(s);
}

// ---------------------------------------------------------------------------
extern "C" void kernel_launch(void* const* d_in, const int* in_sizes, int n_in,
                              void* d_out, int out_size, void* d_ws,
                              size_t ws_size, hipStream_t stream) {
  const float* h      = (const float*)d_in[0];
  const float* fc1_g  = (const float*)d_in[1];
  const float* fc1_b  = (const float*)d_in[2];
  const float* fc1_w  = (const float*)d_in[3];
  const float* fc1_cb = (const float*)d_in[4];
  const float* fc1_a  = (const float*)d_in[5];
  const float* fbl_g  = (const float*)d_in[6];
  const float* fbl_b  = (const float*)d_in[7];
  const float* red_w  = (const float*)d_in[8];
  const float* red_b  = (const float*)d_in[9];
  const float* fW     = (const float*)d_in[10];
  const float* fB     = (const float*)d_in[11];
  const float* exp_w  = (const float*)d_in[12];
  const float* exp_b  = (const float*)d_in[13];
  const float* fc2_g  = (const float*)d_in[14];
  const float* fc2_b  = (const float*)d_in[15];
  const float* fc2_w  = (const float*)d_in[16];
  const float* fc2_cb = (const float*)d_in[17];
  const float* fc2_a  = (const float*)d_in[18];
  float* out = (float*)d_out;

  uint8_t* ws = (uint8_t*)d_ws;
  size_t off = 0;
  auto carve = [&](size_t bytes) -> void* {
    void* p = ws + off;
    off += (bytes + 255) & ~(size_t)255;
    return p;
  };
  float*    x1     = (float*)carve((size_t)NB * FF * CC * sizeof(float));
  _Float16* hpA    = (_Float16*)carve((size_t)CPPC * NPAD * sizeof(_Float16));
  float*    hp2    = (float*)carve((size_t)NB * FF * CPPC * sizeof(float));
  _Float16* fW16   = (_Float16*)carve((size_t)CPPC * FP * FP * sizeof(_Float16));
  _Float16* redw16 = (_Float16*)carve((size_t)CPPC * CC * sizeof(_Float16));
  (void)ws_size; (void)n_in; (void)in_sizes; (void)out_size;

  {
    size_t total = (size_t)CPPC * FP * FP;
    int blocks = (int)((total + 255) / 256);
    prep_kernel<<<blocks, 256, 0, stream>>>(fW, red_w, fW16, redw16);
  }
  freq_conv_kernel<<<dim3(NB, 5), 256, 0, stream>>>(h, x1, fc1_g, fc1_b, fc1_w,
                                                    fc1_cb, fc1_a, 1, 0);
  reduce_kernel<<<dim3(NB, FP / 16), 32, 0, stream>>>(x1, fbl_g, fbl_b, redw16,
                                                      red_b, hpA);
  band_gemm_kernel<<<dim3(NB / 16, FP / 32, CPPC), 32, 0, stream>>>(hpA, fW16,
                                                                    fB, hp2);
  {
    size_t total = (size_t)NB * FF * CC;
    int blocks = (int)((total + 255) / 256);
    expand_kernel<<<blocks, 256, 0, stream>>>(hp2, exp_w, exp_b, x1);
  }
  freq_conv_kernel<<<dim3(NB, 5), 256, 0, stream>>>(x1, out, fc2_g, fc2_b,
                                                    fc2_w, fc2_cb, fc2_a, 0, 1);
}
